// WorldGraphEncoder_22660247453747
// MI455X (gfx1250) — matrix operand order repaired and checked by reading, not compiled, not run on hardware
//
#include <hip/hip_runtime.h>
#include <hip/hip_bf16.h>
#include <cstdint>

typedef __attribute__((ext_vector_type(16))) _Float16 v16h;
typedef __attribute__((ext_vector_type(8)))  float    v8f;

static constexpr int N_NODESC = 100000;
static constexpr int N_EDGESC = 1600000;
static constexpr int E_TOT    = N_EDGESC + N_NODESC;   // self-loops appended
static constexpr int FEATC    = 64;
static constexpr int HIDC     = 128;
static constexpr int HEADSC   = 4;
static constexpr int OUTC     = 128;

#define DEVFN __device__ __forceinline__

DEVFN void atomAddF(float* p, float v) {
    __hip_atomic_fetch_add(p, v, __ATOMIC_RELAXED, __HIP_MEMORY_SCOPE_AGENT);
}

// ordered-uint encoding so atomicMax(u32) implements float max (all reals)
DEVFN unsigned encF(float f) {
    unsigned u = __float_as_uint(f);
    return (u & 0x80000000u) ? ~u : (u | 0x80000000u);
}
DEVFN float decF(unsigned u) {
    u = (u & 0x80000000u) ? (u & 0x7FFFFFFFu) : ~u;
    return __uint_as_float(u);
}

DEVFN void edgeIds(const int* __restrict__ ei, int e, int& s, int& d) {
    if (e < N_EDGESC) { s = ei[e]; d = ei[N_EDGESC + e]; }
    else              { s = e - N_EDGESC; d = s; }
}

// ---------------------------------------------------------------------------
// Tiled GEMM, f32 in/out, f16 WMMA 16x16x32 compute.
// C[M, NT*128] = act(A[M,K] @ B[K, NT*128] + bias)
// Block = 8 waves = one 16-row tile. Wave w owns col tiles {w, w+8, ...} (NT).
// A row-tile staged in LDS via b128; A fragment built once per K-step and
// reused across the NT column tiles (NT WMMAs per K-step).
// ---------------------------------------------------------------------------
template <int K, int NT, bool HAS_BIAS, bool ACT>
__global__ void __launch_bounds__(256)
gemm_wmma_f16_k(const float* __restrict__ A, const float* __restrict__ B,
                const float* __restrict__ bias, float* __restrict__ C)
{
    constexpr int NCOLS = NT * 128;
    __shared__ __align__(16) float ldsA[16 * K];
    const int rt   = blockIdx.x;                     // 16-row tile index
    const int tid  = threadIdx.x;
    const int w    = tid >> 5;
    const int lane = tid & 31;

    {   // 16xK A tile is contiguous in memory -> b128 cooperative stage
        const float4* src = (const float4*)(A + (size_t)rt * 16 * K);
        float4* dst = (float4*)ldsA;
        for (int i = tid; i < 4 * K; i += 256) dst[i] = src[i];
    }
    __syncthreads();

    const int m = lane & 15;
    const int g = lane >> 4;

    v8f acc[NT];
#pragma unroll
    for (int i = 0; i < NT; ++i) acc[i] = (v8f){};

#pragma unroll
    for (int kb = 0; kb < K; kb += 32) {
        // A 16x32 f16 fragment (CDNA5 layout): built once, reused for all NT tiles
        v16h af;
#pragma unroll
        for (int j = 0; j < 8; ++j) {
            const int ka = kb + ((j >> 2) << 4) + ((j & 3) << 1) + (g << 3);
            const float2 a2 = *(const float2*)&ldsA[m * K + ka];
            af[2 * j]     = (_Float16)a2.x;
            af[2 * j + 1] = (_Float16)a2.y;
        }
#pragma unroll
        for (int i = 0; i < NT; ++i) {
            const int n = ((w + i * 8) << 4) + (lane & 15);
            if (kb + 32 < K)
                __builtin_prefetch(&B[(size_t)(kb + 32) * NCOLS + n], 0, 3);
            // B 32x16 fragment: lanes0-15 K=0..7,16..23, lanes16-31 K=8..15,24..31 (2/VGPR)
            v16h bf;
#pragma unroll
            for (int j = 0; j < 8; ++j) {
                const int kbn = kb + ((j >> 2) << 4) + ((j & 3) << 1) + (g << 3);
                bf[2 * j]     = (_Float16)B[(size_t)kbn * NCOLS + n];
                bf[2 * j + 1] = (_Float16)B[((size_t)kbn + 1) * NCOLS + n];
            }
            acc[i] = __builtin_amdgcn_wmma_f32_16x16x32_f16(false, af, false, bf,
                                                            (short)0, acc[i], false, false);
        }
    }

#pragma unroll
    for (int i = 0; i < NT; ++i) {
        const int col = ((w + i * 8) << 4) + (lane & 15);
        const float bv = HAS_BIAS ? bias[col] : 0.0f;
#pragma unroll
        for (int r = 0; r < 8; ++r) {                 // C layout: vgpr r -> M = r + 8*g
            const int row = rt * 16 + r + (g << 3);
            float v = acc[i][r] + bv;
            if (ACT) v = fmaxf(v, 0.0f);
            C[(size_t)row * NCOLS + col] = v;
        }
    }
}

// ---------------------------------------------------------------------------
// per-(node,head) attention coefficients: a = sum_c t[n,hd,c]*att[hd,c]
// ---------------------------------------------------------------------------
__global__ void alpha_coef_k(const float* __restrict__ t,
                             const float* __restrict__ att_s,
                             const float* __restrict__ att_d,
                             float* __restrict__ as_, float* __restrict__ ad_,
                             int C)
{
    const int i = blockIdx.x * blockDim.x + threadIdx.x;
    if (i >= N_NODESC * HEADSC) return;
    const int node = i >> 2, hd = i & 3;
    const float* row = t + (size_t)node * (HEADSC * C) + hd * C;
    const float* sv = att_s + hd * C;
    const float* dv = att_d + hd * C;
    float accs = 0.f, accd = 0.f;
    for (int c = 0; c < C; ++c) { const float v = row[c]; accs += v * sv[c]; accd += v * dv[c]; }
    as_[i] = accs; ad_[i] = accd;
}

__global__ void fill_f32_k(float* p, float v, int n) {
    const int i = blockIdx.x * 256 + threadIdx.x;
    if (i < n) p[i] = v;
}
__global__ void fill_u32_k(unsigned* p, unsigned v, int n) {
    const int i = blockIdx.x * 256 + threadIdx.x;
    if (i < n) p[i] = v;
}

__global__ void edge_max_k(const int* __restrict__ ei,
                           const float* __restrict__ as_, const float* __restrict__ ad_,
                           unsigned* __restrict__ emax)
{
    const int e = blockIdx.x * 256 + threadIdx.x;
    if (e >= E_TOT) return;
    int s, d; edgeIds(ei, e, s, d);
#pragma unroll
    for (int hd = 0; hd < HEADSC; ++hd) {
        float v = as_[s * 4 + hd] + ad_[d * 4 + hd];
        v = v > 0.f ? v : 0.2f * v;                   // leaky_relu(0.2)
        atomicMax(&emax[d * 4 + hd], encF(v));
    }
}

__global__ void edge_den_k(const int* __restrict__ ei,
                           const float* __restrict__ as_, const float* __restrict__ ad_,
                           const unsigned* __restrict__ emax, float* __restrict__ den)
{
    const int e = blockIdx.x * 256 + threadIdx.x;
    if (e >= E_TOT) return;
    int s, d; edgeIds(ei, e, s, d);
#pragma unroll
    for (int hd = 0; hd < HEADSC; ++hd) {
        float v = as_[s * 4 + hd] + ad_[d * 4 + hd];
        v = v > 0.f ? v : 0.2f * v;
        atomAddF(&den[d * 4 + hd], __expf(v - decF(emax[d * 4 + hd])));
    }
}

// weighted scatter-aggregate; one thread per (edge, head)
__global__ void edge_agg_k(const int* __restrict__ ei,
                           const float* __restrict__ as_, const float* __restrict__ ad_,
                           const unsigned* __restrict__ emax, const float* __restrict__ den,
                           const float* __restrict__ t, float* __restrict__ out,
                           int C, int strideT, int concat)
{
    const int tid = blockIdx.x * 256 + threadIdx.x;
    if (tid >= E_TOT * HEADSC) return;
    const int e = tid >> 2, hd = tid & 3;
    int s, d; edgeIds(ei, e, s, d);
    float v = as_[s * 4 + hd] + ad_[d * 4 + hd];
    v = v > 0.f ? v : 0.2f * v;
    const float ex = __expf(v - decF(emax[d * 4 + hd]));
    const float alpha = ex / den[d * 4 + hd];
    const float* trow = t + (size_t)s * strideT + hd * C;
    if (concat) {
        float* orow = out + (size_t)d * HIDC + hd * C;
#pragma unroll 4
        for (int c = 0; c < C; ++c) atomAddF(&orow[c], alpha * trow[c]);
    } else {
        float* orow = out + (size_t)d * OUTC;
        const float sc = alpha * 0.25f;               // mean over 4 heads
#pragma unroll 4
        for (int c = 0; c < C; ++c) atomAddF(&orow[c], sc * trow[c]);
    }
}

__global__ void bn_elu_k(const float* __restrict__ o1, const float* __restrict__ b1,
                         const float* __restrict__ gamma, const float* __restrict__ beta,
                         float* __restrict__ h1, int n)
{
    const int i = blockIdx.x * 256 + threadIdx.x;
    if (i >= n) return;
    const int c = i & (HIDC - 1);
    const float x = (o1[i] + b1[c]) * (gamma[c] * rsqrtf(1.0f + 1e-5f)) + beta[c];
    h1[i] = x > 0.f ? x : (__expf(x) - 1.0f);        // ELU
}

__global__ void colsum_k(const float* __restrict__ h, float* __restrict__ gsum)
{
    const int c = threadIdx.x;                        // 128 threads
    const int rows = (N_NODESC + gridDim.x - 1) / gridDim.x;
    const int r0 = blockIdx.x * rows;
    const int r1 = min(N_NODESC, r0 + rows);
    float acc = 0.f;
    for (int r = r0; r < r1; ++r) acc += h[(size_t)r * OUTC + c];
    atomAddF(&gsum[c], acc);
}

__global__ void final_k(const float* __restrict__ gsum, const float* __restrict__ b2,
                        const float* __restrict__ Wo, const float* __restrict__ bo,
                        float* __restrict__ out)
{
    __shared__ float g[OUTC];
    const int c = threadIdx.x;
    g[c] = gsum[c] * (1.0f / (float)N_NODESC) + b2[c];
    __syncthreads();
    float acc = bo[c];
    for (int k = 0; k < OUTC; ++k) acc += g[k] * Wo[k * OUTC + c];
    out[c] = fmaxf(acc, 0.0f);
}

// ---------------------------------------------------------------------------
extern "C" void kernel_launch(void* const* d_in, const int* in_sizes, int n_in,
                              void* d_out, int out_size, void* d_ws, size_t ws_size,
                              hipStream_t stream)
{
    const float* x        = (const float*)d_in[0];
    const int*   ei       = (const int*)d_in[1];     // edge_index [2,E] (harness int ABI)
    const float* Wp       = (const float*)d_in[2];
    const float* bp       = (const float*)d_in[3];
    const float* W1       = (const float*)d_in[4];
    const float* att_src1 = (const float*)d_in[5];
    const float* att_dst1 = (const float*)d_in[6];
    const float* b1       = (const float*)d_in[7];
    const float* gamma    = (const float*)d_in[8];
    const float* beta     = (const float*)d_in[9];
    const float* W2       = (const float*)d_in[10];
    const float* att_src2 = (const float*)d_in[11];
    const float* att_dst2 = (const float*)d_in[12];
    const float* b2       = (const float*)d_in[13];
    const float* Wo       = (const float*)d_in[14];
    const float* bo       = (const float*)d_in[15];
    float* out = (float*)d_out;

    // workspace layout (buffer reuse: buf0 = h0 -> o1 -> h2, buf1 = t1 -> h1)
    float*    buf0 = (float*)d_ws;                                // [N,128]
    float*    buf1 = buf0 + (size_t)N_NODESC * HIDC;              // [N,128]
    float*    buf2 = buf1 + (size_t)N_NODESC * HIDC;              // [N,512] t2
    float*    as_  = buf2 + (size_t)N_NODESC * HEADSC * OUTC;     // [N,4]
    float*    ad_  = as_  + (size_t)N_NODESC * HEADSC;            // [N,4]
    unsigned* emax = (unsigned*)(ad_ + (size_t)N_NODESC * HEADSC);// [N,4]
    float*    den  = (float*)(emax + (size_t)N_NODESC * HEADSC);  // [N,4]
    float*    gsum = den + (size_t)N_NODESC * HEADSC;             // [128]

    const int mtiles = N_NODESC / 16;                             // 6250 exact
    const int gNH    = (N_NODESC * HEADSC + 255) / 256;
    const int gE     = (E_TOT + 255) / 256;
    const int gEH    = (E_TOT * HEADSC + 255) / 256;
    const int gN128  = (N_NODESC * HIDC + 255) / 256;

    // h0 = relu(x @ Wp + bp)            -> buf0
    gemm_wmma_f16_k<64, 1, true, true><<<mtiles, 256, 0, stream>>>(x, Wp, bp, buf0);
    // t1 = h0 @ W1                      -> buf1
    gemm_wmma_f16_k<128, 1, false, false><<<mtiles, 256, 0, stream>>>(buf0, W1, nullptr, buf1);
    // attention coefficients, layer 1 (C=32)
    alpha_coef_k<<<gNH, 256, 0, stream>>>(buf1, att_src1, att_dst1, as_, ad_, HIDC / HEADSC);
    // init segment buffers + o1
    fill_u32_k<<<gNH, 256, 0, stream>>>(emax, 0u, N_NODESC * HEADSC);
    fill_f32_k<<<gNH, 256, 0, stream>>>(den, 0.f, N_NODESC * HEADSC);
    fill_f32_k<<<gN128, 256, 0, stream>>>(buf0, 0.f, N_NODESC * HIDC);
    // segment softmax + aggregate (concat heads) -> buf0 (o1)
    edge_max_k<<<gE, 256, 0, stream>>>(ei, as_, ad_, emax);
    edge_den_k<<<gE, 256, 0, stream>>>(ei, as_, ad_, emax, den);
    edge_agg_k<<<gEH, 256, 0, stream>>>(ei, as_, ad_, emax, den, buf1, buf0,
                                        HIDC / HEADSC, HIDC, 1);
    // h1 = elu(bn(o1 + b1))             -> buf1
    bn_elu_k<<<gN128, 256, 0, stream>>>(buf0, b1, gamma, beta, buf1, N_NODESC * HIDC);
    // t2 = h1 @ W2                      -> buf2  [N, 4*128]
    gemm_wmma_f16_k<128, 4, false, false><<<mtiles, 256, 0, stream>>>(buf1, W2, nullptr, buf2);
    // attention coefficients, layer 2 (C=128)
    alpha_coef_k<<<gNH, 256, 0, stream>>>(buf2, att_src2, att_dst2, as_, ad_, OUTC);
    // re-init segment buffers + h2 + gsum
    fill_u32_k<<<gNH, 256, 0, stream>>>(emax, 0u, N_NODESC * HEADSC);
    fill_f32_k<<<gNH, 256, 0, stream>>>(den, 0.f, N_NODESC * HEADSC);
    fill_f32_k<<<gN128, 256, 0, stream>>>(buf0, 0.f, N_NODESC * HIDC);
    fill_f32_k<<<1, 256, 0, stream>>>(gsum, 0.f, OUTC);
    // segment softmax + aggregate (mean heads) -> buf0 (h2); scatter target fits in L2
    edge_max_k<<<gE, 256, 0, stream>>>(ei, as_, ad_, emax);
    edge_den_k<<<gE, 256, 0, stream>>>(ei, as_, ad_, emax, den);
    edge_agg_k<<<gEH, 256, 0, stream>>>(ei, as_, ad_, emax, den, buf2, buf0,
                                        OUTC, HEADSC * OUTC, 0);
    // graph mean + final relu(g @ Wo + bo)
    colsum_k<<<400, 128, 0, stream>>>(buf0, gsum);
    final_k<<<1, 128, 0, stream>>>(gsum, b2, Wo, bo, out);
}
